// OptimizedSparseKeypointDetector_4277787426928
// MI455X (gfx1250) — compile-verified, matrix operand order checked by reading, not run on hardware
//
#include <hip/hip_runtime.h>
#include <stdint.h>

// ---------------- problem constants ----------------
#define Bsz   16
#define Hd    512
#define Wd    1024
#define MAXK  512
#define RR    3
#define NCAND 2048
#define CAP   16384          // per-image candidate capacity (expected ~10K)

// ---------------- workspace layout ----------------
#define ADJ_WORDS 64                                   // 2048 bits per row
#define WS_COUNT_OFF 0
#define WS_ADJ_OFF   256
#define WS_ADJ_BYTES ((size_t)Bsz * NCAND * ADJ_WORDS * 4)     // 8 MiB
#define WS_SCORE_OFF (WS_ADJ_OFF + WS_ADJ_BYTES)
#define WS_IDX_OFF   (WS_SCORE_OFF + (size_t)Bsz * CAP * 4)

typedef __attribute__((ext_vector_type(16))) _Float16 v16h;
typedef __attribute__((ext_vector_type(8)))  float    v8f;
typedef __attribute__((ext_vector_type(4)))  unsigned int v4u;
typedef __attribute__((ext_vector_type(8)))  int      v8i;
typedef __attribute__((ext_vector_type(4)))  int      v4i;

#if __has_builtin(__builtin_amdgcn_tensor_load_to_lds) && \
    __has_builtin(__builtin_amdgcn_s_wait_tensorcnt)
#define HAVE_TDM 1
#endif

// =====================================================================
// Kernel 1: streaming peak detect (memory-bound; ~64 MiB total traffic)
// =====================================================================
#define T_W  64
#define T_H  16
#define HALO 3
#define SH_W (T_W + 2 * HALO)   // 70
#define SH_H (T_H + 2 * HALO)   // 22
#define SH_WP 72                // padded to dodge LDS bank conflicts

__global__ __launch_bounds__(256)
void peak_detect_kernel(const float* __restrict__ gray,
                        const float* __restrict__ mask,
                        float* __restrict__ cand_sc,
                        int*   __restrict__ cand_ix,
                        int*   __restrict__ cand_cnt)
{
    __shared__ float s_prod[SH_H][SH_WP];
    __shared__ float s_rmax[SH_H][T_W];
    __shared__ float s_mask[T_H][T_W];

    const int b   = blockIdx.z;
    const int x0  = blockIdx.x * T_W;
    const int y0  = blockIdx.y * T_H;
    const int tid = threadIdx.x;
    const size_t plane = (size_t)b * Hd * Wd;
    const float* g = gray + plane;
    const float* m = mask + plane;

    // gfx1250 prefetch path for the next tile row
    if (tid == 0) {
        int py = y0 + T_H < Hd ? y0 + T_H : y0;
        __builtin_prefetch(g + (size_t)py * Wd + x0, 0, 1);
        __builtin_prefetch(m + (size_t)py * Wd + x0, 0, 1);
    }

    // load halo tile of products (OOB -> -inf, matching reduce_window init)
    for (int i = tid; i < SH_H * SH_W; i += 256) {
        int r = i / SH_W, c = i % SH_W;
        int gy = y0 - HALO + r, gx = x0 - HALO + c;
        float p = -INFINITY;
        if (gx >= 0 && gx < Wd && gy >= 0 && gy < Hd) {
            float gv = g[(size_t)gy * Wd + gx];
            float mv = m[(size_t)gy * Wd + gx];
            p = gv * mv;
            if ((unsigned)(r - HALO) < (unsigned)T_H &&
                (unsigned)(c - HALO) < (unsigned)T_W)
                s_mask[r - HALO][c - HALO] = mv;
        }
        s_prod[r][c] = p;
    }
    __syncthreads();

    // separable 7x7 max: horizontal pass
    for (int i = tid; i < SH_H * T_W; i += 256) {
        int r = i / T_W, c = i % T_W;
        float v = s_prod[r][c];
#pragma unroll
        for (int k = 1; k < 7; ++k) v = fmaxf(v, s_prod[r][c + k]);
        s_rmax[r][c] = v;
    }
    __syncthreads();

    // vertical pass + candidate test + append
    for (int i = tid; i < T_H * T_W; i += 256) {
        int oy = i / T_W, ox = i % T_W;
        float v = s_rmax[oy][ox];
#pragma unroll
        for (int k = 1; k < 7; ++k) v = fmaxf(v, s_rmax[oy + k][ox]);
        float p = s_prod[oy + HALO][ox + HALO];
        if (p == v && p > 0.1f && s_mask[oy][ox] > 0.0f) {
            int c = atomicAdd(&cand_cnt[b], 1);
            if (c < CAP) {
                cand_sc[(size_t)b * CAP + c] = p;
                cand_ix[(size_t)b * CAP + c] = (y0 + oy) * Wd + (x0 + ox);
            }
        }
    }
}

// =====================================================================
// Kernel 2: per-image top-2048 select + sort + WMMA adjacency + NMS
// one workgroup (32 waves) per image
// =====================================================================
__global__ __launch_bounds__(1024)
void select_nms_kernel(const float* __restrict__ cand_sc,
                       const int*   __restrict__ cand_ix,
                       const int*   __restrict__ cand_cnt,
                       unsigned*    __restrict__ adj_all,
                       float*       __restrict__ out)
{
    extern __shared__ __align__(16) unsigned char smem[];
    float* s_sc = (float*)smem;                                   // [CAP] 64KB (phase A)
    float* s_x  = (float*)smem;                                   // [2048] (reuse)
    float* s_y  = (float*)(smem + 8192);
    float* s_n  = (float*)(smem + 16384);
    float* s_v  = (float*)(smem + 24576);                         // score or 0
    unsigned long long* s_key = (unsigned long long*)(smem + 65536); // [2048] 16KB
    unsigned* s_misc = (unsigned*)(smem + 65536 + 16384);
    // s_misc: [0..31] warp partials, [32] total, [33] cntG, [34] cntE,
    //         [64..127] keepA, [128..191] keepB

    const int b    = blockIdx.x;
    const int tid  = threadIdx.x;
    const int lane = tid & 31;
    const int warp = tid >> 5;

    float* kp_out = out + (size_t)b * MAXK * 2;
    float* sc_out = out + (size_t)Bsz * MAXK * 2 + (size_t)b * MAXK;
    if (tid < MAXK) {                       // d_out is poisoned: zero-fill
        kp_out[2 * tid]     = 0.0f;
        kp_out[2 * tid + 1] = 0.0f;
        sc_out[tid]         = 0.0f;
    }

    const int n = cand_cnt[b] < CAP ? cand_cnt[b] : CAP;
    const float* csc = cand_sc + (size_t)b * CAP;
    const int*   cix = cand_ix + (size_t)b * CAP;

#ifdef HAVE_TDM
    // Tensor Data Mover: one 1-D descriptor DMAs the whole 64KB candidate
    // score block into LDS (dynamic-LDS byte offset = groupstaticsize()).
    if (warp == 0) {
        unsigned long long ga = (unsigned long long)(uintptr_t)csc;
        unsigned lds_off = __builtin_amdgcn_groupstaticsize();
        v4u g0 = { 1u,                                    // count=1 (valid D#)
                   lds_off,                               // lds_addr
                   (unsigned)ga,                          // global_addr[31:0]
                   (unsigned)(ga >> 32) | (2u << 30) };   // addr[56:32] | type=2
        v8i g1 = { (int)(2u << 16),          // data_size = 4B
                   (int)(16384u << 16),      // tensor_dim0[15:0] @ bits 63:48
                   (int)(1u << 16),          // tensor_dim0 hi=0, tensor_dim1=1
                   (int)(16384u << 16),      // tile_dim0 = 16384 @ bits 127:112
                   0,                        // tile_dim1/2 unused
                   16384, 0, 0 };            // tensor_dim0_stride = 16384
        v4i gz = { 0, 0, 0, 0 };
#if defined(__clang_major__) && (__clang_major__ >= 23)
        v8i gz8 = { 0, 0, 0, 0, 0, 0, 0, 0 };
        __builtin_amdgcn_tensor_load_to_lds(g0, g1, gz, gz, gz8, 0);
#else
        __builtin_amdgcn_tensor_load_to_lds(g0, g1, gz, gz, 0);
#endif
        __builtin_amdgcn_s_wait_tensorcnt(0);
    }
    __syncthreads();
#else
    for (int i = tid; i < n; i += 1024) s_sc[i] = csc[i];
    __syncthreads();
#endif

    // ---- phase A: exact 2048-th score threshold via bitwise binary search ----
    auto count_ge = [&](unsigned t) -> unsigned {
        unsigned c = 0;
        for (int i = tid; i < n; i += 1024)
            c += (__float_as_uint(s_sc[i]) >= t) ? 1u : 0u;
        for (int d = 16; d > 0; d >>= 1) c += __shfl_down(c, d, 32);
        if (lane == 0) s_misc[warp] = c;
        __syncthreads();
        if (warp == 0) {
            unsigned v = s_misc[lane];
            for (int d = 16; d > 0; d >>= 1) v += __shfl_down(v, d, 32);
            if (lane == 0) s_misc[32] = v;
        }
        __syncthreads();
        unsigned tot = s_misc[32];
        __syncthreads();
        return tot;
    };

    unsigned lo = 0;
    for (int bit = 31; bit >= 0; --bit) {
        unsigned t = lo | (1u << bit);
        if (count_ge(t) >= (unsigned)NCAND) lo = t;
    }

    // ---- gather top-2048 as packed keys (score<<32 | ~idx) ----
    if (tid == 0) { s_misc[33] = 0; s_misc[34] = 0; }
    __syncthreads();
    for (int i = tid; i < n; i += 1024) {
        unsigned sb = __float_as_uint(s_sc[i]);
        if (sb > lo) {
            unsigned p = atomicAdd(&s_misc[33], 1u);
            s_key[p] = ((unsigned long long)sb << 32) | (unsigned)(~(unsigned)cix[i]);
        }
    }
    __syncthreads();
    unsigned cntG = s_misc[33];
    for (int i = tid; i < n; i += 1024) {
        unsigned sb = __float_as_uint(s_sc[i]);
        if (sb == lo) {
            unsigned p = atomicAdd(&s_misc[34], 1u) + cntG;
            if (p < (unsigned)NCAND)
                s_key[p] = ((unsigned long long)sb << 32) | (unsigned)(~(unsigned)cix[i]);
        }
    }
    __syncthreads();
    int filled = (int)(cntG + s_misc[34]);
    if (filled > NCAND) filled = NCAND;
    for (int i = filled + tid; i < NCAND; i += 1024) s_key[i] = 0ull;
    __syncthreads();

    // ---- bitonic sort 2048 keys, descending (score desc, idx asc) ----
    for (unsigned k = 2; k <= (unsigned)NCAND; k <<= 1) {
        for (unsigned j = k >> 1; j > 0; j >>= 1) {
#pragma unroll
            for (int h = 0; h < 2; ++h) {
                unsigned t = (unsigned)tid + (unsigned)h * 1024u;
                unsigned x = t ^ j;
                if (x > t) {
                    unsigned long long a = s_key[t], c = s_key[x];
                    bool descBlk = ((t & k) == 0);
                    bool sw = descBlk ? (a < c) : (a > c);
                    if (sw) { s_key[t] = c; s_key[x] = a; }
                }
            }
            __syncthreads();
        }
    }

    // ---- unpack coords / norms / scores ----
    for (int i = tid; i < NCAND; i += 1024) {
        unsigned long long kk = s_key[i];
        float sc = __uint_as_float((unsigned)(kk >> 32));
        unsigned idx = ~(unsigned)kk;
        float x = (float)(idx & (Wd - 1));
        float y = (float)(idx >> 10);
        bool val = (sc > 0.1f) && (idx < (unsigned)(Wd * Hd));
        s_x[i] = x;
        s_y[i] = y;
        s_n[i] = x * x + y * y;
        s_v[i] = val ? sc : 0.0f;
    }
    __syncthreads();

    // ---- WMMA adjacency: dist^2(i,j) = n_i + n_j - 2 * (P P^T)_ij ----
    // A (16x32 f16): lanes 0-15 carry row M=lane, K=0,1 = (x,y); rest zero.
    // B (32x16 f16): lanes 0-15 carry col N=lane, K=0,1 = (x,y); rest zero.
    unsigned* adj = adj_all + (size_t)b * NCAND * ADJ_WORDS;
    for (int t = warp; t < 128 * 128; t += 32) {
        int ti = t >> 7, tj = t & 127;
        if (tj > ti) continue;               // only j <= i halves are consumed
        int i0 = ti << 4, j0 = tj << 4;
        bool lowhalf = lane < 16;
        int li = i0 + (lane & 15);
        int lj = j0 + (lane & 15);
        v16h a = {};
        v16h bb = {};
        if (lowhalf) {
            a[0]  = (_Float16)s_x[li];  a[1]  = (_Float16)s_y[li];
            bb[0] = (_Float16)s_x[lj];  bb[1] = (_Float16)s_y[lj];
        }
        v8f cc = {};
        v8f d = __builtin_amdgcn_wmma_f32_16x16x32_f16(
            false, a, false, bb, (short)0, cc, false, false);

        int   jg = j0 + (lane & 15);
        float nj = s_n[jg];
#pragma unroll
        for (int e = 0; e < 8; ++e) {
            int   ig = i0 + e + (lowhalf ? 0 : 8);
            float d2 = s_n[ig] + nj - 2.0f * d[e];
            bool hit = (d2 < (float)(RR * RR)) && (ig != jg);
            unsigned msk = __builtin_amdgcn_ballot_w32(hit);
            if (lane == 0) {
                unsigned w  = (unsigned)(j0 >> 5);
                unsigned sh = (unsigned)(j0 & 16);
                atomicOr(&adj[(size_t)(i0 + e) * ADJ_WORDS + w],
                         (msk & 0xFFFFu) << sh);
                atomicOr(&adj[(size_t)(i0 + e + 8) * ADJ_WORDS + w],
                         (msk >> 16) << sh);
            }
        }
    }
    __threadfence();
    __syncthreads();

    // ---- greedy NMS as Jacobi relaxation (converges to sequential result) ----
    unsigned* keepA = s_misc + 64;
    unsigned* keepB = s_misc + 128;
    if (tid < ADJ_WORDS) {
        unsigned w = 0;
        for (int q = 0; q < 32; ++q)
            if (s_v[tid * 32 + q] > 0.1f) w |= (1u << q);
        keepA[tid] = w;
    }
    __syncthreads();

    for (int it = 0; it < 12; ++it) {
        if (tid < ADJ_WORDS) keepB[tid] = 0u;
        __syncthreads();
#pragma unroll
        for (int h = 0; h < 2; ++h) {
            int i = tid + h * 1024;
            bool v = s_v[i] > 0.1f;
            bool nk = false;
            if (v) {
                const unsigned* row = &adj[(size_t)i * ADJ_WORDS];
                unsigned sup = 0;
                int wl = i >> 5;
                for (int w = 0; w < wl; ++w) sup |= (row[w] & keepA[w]);
                unsigned mlow = (i & 31) ? ((1u << (i & 31)) - 1u) : 0u;
                sup |= (row[wl] & keepA[wl] & mlow);
                nk = (sup == 0u);
            }
            if (nk) atomicOr(&keepB[i >> 5], 1u << (i & 31));
        }
        __syncthreads();
        if (tid < ADJ_WORDS) keepA[tid] = keepB[tid];
        __syncthreads();
    }

    // ---- compaction: warp-scan prefix sum over keep bits, write outputs ----
    int i0e = 2 * tid, i1e = 2 * tid + 1;
    unsigned k0 = (keepA[i0e >> 5] >> (i0e & 31)) & 1u;
    unsigned k1 = (keepA[i1e >> 5] >> (i1e & 31)) & 1u;
    unsigned local = k0 + k1;
    unsigned incl = local;
    for (int dd = 1; dd < 32; dd <<= 1) {
        unsigned tu = __shfl_up(incl, dd, 32);
        if (lane >= dd) incl += tu;
    }
    if (lane == 31) s_misc[warp] = incl;
    __syncthreads();
    if (warp == 0) {
        unsigned v = s_misc[lane];
        unsigned iv = v;
        for (int dd = 1; dd < 32; dd <<= 1) {
            unsigned tu = __shfl_up(iv, dd, 32);
            if (lane >= dd) iv += tu;
        }
        s_misc[lane] = iv - v;     // exclusive warp base
    }
    __syncthreads();
    unsigned base = s_misc[warp] + (incl - local);
    unsigned p0 = base, p1 = base + k0;
    if (k0 && p0 < MAXK) {
        kp_out[2 * p0] = s_x[i0e]; kp_out[2 * p0 + 1] = s_y[i0e];
        sc_out[p0] = s_v[i0e];
    }
    if (k1 && p1 < MAXK) {
        kp_out[2 * p1] = s_x[i1e]; kp_out[2 * p1 + 1] = s_y[i1e];
        sc_out[p1] = s_v[i1e];
    }
}

// =====================================================================
extern "C" void kernel_launch(void* const* d_in, const int* in_sizes, int n_in,
                              void* d_out, int out_size, void* d_ws, size_t ws_size,
                              hipStream_t stream)
{
    (void)in_sizes; (void)n_in; (void)out_size; (void)ws_size;
    const float* gray = (const float*)d_in[0];
    const float* mask = (const float*)d_in[1];
    float* out = (float*)d_out;
    unsigned char* ws = (unsigned char*)d_ws;

    int*      cnt = (int*)(ws + WS_COUNT_OFF);
    unsigned* adj = (unsigned*)(ws + WS_ADJ_OFF);
    float*    csc = (float*)(ws + WS_SCORE_OFF);
    int*      cix = (int*)(ws + WS_IDX_OFF);

    // zero candidate counters + adjacency bitmasks (graph-capturable)
    hipMemsetAsync(ws, 0, WS_ADJ_OFF + WS_ADJ_BYTES, stream);

    dim3 g1(Wd / T_W, Hd / T_H, Bsz);
    peak_detect_kernel<<<g1, 256, 0, stream>>>(gray, mask, csc, cix, cnt);

    size_t shmem = 65536 + 16384 + 2048;   // 82 KB dynamic LDS (<< 320 KB/WGP)
    select_nms_kernel<<<Bsz, 1024, shmem, stream>>>(csc, cix, cnt, adj, out);
}
